// HierarchicalMoE_53068615909669
// MI455X (gfx1250) — matrix operand order repaired and checked by reading, not compiled, not run on hardware
//
#include <hip/hip_runtime.h>
#include <math.h>

typedef __attribute__((ext_vector_type(16))) __bf16 v16bf;
typedef __attribute__((ext_vector_type(8)))  float  v8f;
typedef unsigned short u16;
typedef unsigned int   u32;

static constexpr int T_TOK = 8192;   // B*L
static constexpr int DM    = 512;    // d_model
static constexpr int NE    = 8;      // experts
static constexpr int DFF   = 2048;   // d_ff
static constexpr int NR    = T_TOK * 2;  // assignment rows (token,slot)
static constexpr float LBW = 0.01f;

__device__ __forceinline__ u16 f2bf(float f) {
  union { float f; u32 u; } x; x.f = f;
  u32 u = x.u;
  // round-to-nearest-even (inputs are finite)
  u32 r = (u + 0x7FFFu + ((u >> 16) & 1u)) >> 16;
  return (u16)r;
}

// ---------------- fp32 -> bf16 conversion (weights) ----------------
__global__ void cvt_bf16_kernel(const float* __restrict__ s, u16* __restrict__ d, int n) {
  int i = (blockIdx.x * blockDim.x + threadIdx.x) * 4;
  if (i + 3 < n) {
    float4 v = *(const float4*)(s + i);
    d[i + 0] = f2bf(v.x); d[i + 1] = f2bf(v.y);
    d[i + 2] = f2bf(v.z); d[i + 3] = f2bf(v.w);
  }
}

__global__ void init_cnt_kernel(int* cnt) {
  if (threadIdx.x < NE) cnt[threadIdx.x] = 0;
}

// ---------------- gating: logits, softmax, top-2, prob partial sums ----------------
__global__ __launch_bounds__(256)
void gate_kernel(const float* __restrict__ x, const float* __restrict__ gw,
                 int* __restrict__ idx, float* __restrict__ score,
                 float* __restrict__ psum /* [gridDim.x * NE] */) {
  __shared__ float sgw[NE * DM];   // 16 KB
  __shared__ float red[256];
  const int tid = threadIdx.x;
  for (int i = tid; i < NE * DM; i += 256) sgw[i] = gw[i];
  __syncthreads();

  const int t = blockIdx.x * 256 + tid;
  float lg[NE];
#pragma unroll
  for (int e = 0; e < NE; e++) lg[e] = 0.f;
  const float* xr = x + (size_t)t * DM;
  for (int k = 0; k < DM; k += 4) {
    float4 xv = *(const float4*)(xr + k);
#pragma unroll
    for (int e = 0; e < NE; e++) {
      lg[e] += xv.x * sgw[e * DM + k + 0] + xv.y * sgw[e * DM + k + 1]
             + xv.z * sgw[e * DM + k + 2] + xv.w * sgw[e * DM + k + 3];
    }
  }
  float m = lg[0];
#pragma unroll
  for (int e = 1; e < NE; e++) m = fmaxf(m, lg[e]);
  float p[NE]; float s = 0.f;
#pragma unroll
  for (int e = 0; e < NE; e++) { p[e] = expf(lg[e] - m); s += p[e]; }
  float inv = 1.f / s;
#pragma unroll
  for (int e = 0; e < NE; e++) p[e] *= inv;

  // top-2, ties -> lowest index first (strict > comparisons)
  int b0 = 0;
#pragma unroll
  for (int e = 1; e < NE; e++) if (p[e] > p[b0]) b0 = e;
  int b1 = (b0 == 0) ? 1 : 0;
#pragma unroll
  for (int e = 0; e < NE; e++) if (e != b1 && e != b0 && p[e] > p[b1]) b1 = e;

  idx[2 * t + 0] = b0;   score[2 * t + 0] = p[b0];
  idx[2 * t + 1] = b1;   score[2 * t + 1] = p[b1];

  // deterministic fixed-order reduction of probs per expert
  for (int e = 0; e < NE; e++) {
    red[tid] = p[e];
    __syncthreads();
    for (int off = 128; off > 0; off >>= 1) {
      if (tid < off) red[tid] += red[tid + off];
      __syncthreads();
    }
    if (tid == 0) psum[blockIdx.x * NE + e] = red[0];
    __syncthreads();
  }
}

// ---------------- routing: compact per-expert lists ----------------
__global__ void route_kernel(const int* __restrict__ idx, int* __restrict__ cnt,
                             int* __restrict__ list) {
  int r = blockIdx.x * 256 + threadIdx.x;   // r = 2*t + slot
  if (r < NR) {
    int e = idx[r];
    int pos = atomicAdd(&cnt[e], 1);
    list[e * T_TOK + pos] = r;
  }
}

// ---------------- aux loss ----------------
__global__ void aux_kernel(const float* __restrict__ psum, const int* __restrict__ cnt,
                           float* __restrict__ outAux, int nblk) {
  if (threadIdx.x == 0 && blockIdx.x == 0) {
    float a = 0.f;
    for (int e = 0; e < NE; e++) {
      float sp = 0.f;
      for (int b = 0; b < nblk; b++) sp += psum[b * NE + e];
      a += ((float)cnt[e] / (float)T_TOK) * (sp / (float)T_TOK);
    }
    *outAux = (float)NE * a * LBW;
  }
}

// ---------------- FFN stage 1: act = silu(x W1^T) * (x V1^T), bf16 WMMA ----------------
// block: 256 thr = 8 waves (4 M-slices x 2 N-slices); tile 64 rows x 128 f-cols
__global__ __launch_bounds__(256)
void ffn1_kernel(const float* __restrict__ x, const u16* __restrict__ w1,
                 const u16* __restrict__ v1, const int* __restrict__ cnt,
                 const int* __restrict__ list, u16* __restrict__ act) {
  const int e = blockIdx.z;
  const int rows = cnt[e];
  const int m0 = blockIdx.x * 64;
  if (m0 >= rows) return;                      // block-uniform: EXEC stays all-1s
  int nrows = rows - m0; if (nrows > 64) nrows = 64;
  const int f0 = blockIdx.y * 128;

  __shared__ __align__(16) u16 sA[64 * 32];    // 4 KB A tile (bf16)
  __shared__ int srow[64];

  const int tid = threadIdx.x, lane = tid & 31, wid = tid >> 5;
  const int wm = wid & 3, wn = wid >> 2;

  if (tid < 64) srow[tid] = (tid < nrows) ? list[e * T_TOK + m0 + tid] : -1;

  v8f ch[4], cv[4];
#pragma unroll
  for (int n = 0; n < 4; n++)
#pragma unroll
    for (int j = 0; j < 8; j++) { ch[n][j] = 0.f; cv[n][j] = 0.f; }

  const u16* w1e = w1 + (size_t)e * DFF * DM;
  const u16* v1e = v1 + (size_t)e * DFF * DM;

  const int arow = tid >> 2;           // 0..63
  const int akq  = (tid & 3) * 8;      // 0,8,16,24

  for (int k0 = 0; k0 < DM; k0 += 32) {
    __syncthreads();
    {
      int code = srow[arow];
      union { u16 h[8]; float4 f; } pk;
      if (code >= 0) {
        const float* src = x + (size_t)(code >> 1) * DM + k0 + akq;
        float4 a0 = *(const float4*)(src);
        float4 a1 = *(const float4*)(src + 4);
        pk.h[0] = f2bf(a0.x); pk.h[1] = f2bf(a0.y); pk.h[2] = f2bf(a0.z); pk.h[3] = f2bf(a0.w);
        pk.h[4] = f2bf(a1.x); pk.h[5] = f2bf(a1.y); pk.h[6] = f2bf(a1.z); pk.h[7] = f2bf(a1.w);
      } else {
        pk.f = make_float4(0.f, 0.f, 0.f, 0.f);
      }
      *(float4*)&sA[arow * 32 + akq] = pk.f;
    }
    __syncthreads();

    // A fragment, contiguous K order matching B (lanes 0-15 -> K..K+15, 16-31 -> K+16..31)
    union { v16bf v; u16 u[16]; } af;
    const int ml = wm * 16 + (lane & 15);
    const int khalf = (lane & 16) ? 8 : 0;
#pragma unroll
    for (int v = 0; v < 8; v++) {
      int kk = 2 * v + 2 * khalf;
      af.u[2 * v + 0] = sA[ml * 32 + kk + 0];
      af.u[2 * v + 1] = sA[ml * 32 + kk + 1];
    }
    // B fragments: lane holds column f; K half-split (lanes 0-15 -> K..K+15, 16-31 -> K+16..31)
    const int fcol = f0 + wn * 64 + (lane & 15);
    const int krow = k0 + ((lane & 16) ? 16 : 0);
#pragma unroll
    for (int n = 0; n < 4; n++) {
      const int f = fcol + n * 16;
      v16bf bw = *(const v16bf*)(w1e + (size_t)f * DM + krow);
      v16bf bv = *(const v16bf*)(v1e + (size_t)f * DM + krow);
      ch[n] = __builtin_amdgcn_wmma_f32_16x16x32_bf16(false, af.v, false, bw,
                                                      (short)0, ch[n], false, false);
      cv[n] = __builtin_amdgcn_wmma_f32_16x16x32_bf16(false, af.v, false, bv,
                                                      (short)0, cv[n], false, false);
    }
  }

  // epilogue: act = silu(h) * v  (C/D layout: VGPR j -> M=j or j+8; lane -> N)
#pragma unroll
  for (int n = 0; n < 4; n++) {
#pragma unroll
    for (int j = 0; j < 8; j++) {
      int m = wm * 16 + j + ((lane & 16) ? 8 : 0);
      if (m < nrows) {
        int f = f0 + wn * 64 + n * 16 + (lane & 15);
        float hh = ch[n][j], vv = cv[n][j];
        float sact = (hh / (1.f + expf(-hh))) * vv;
        int code = srow[m];
        act[(size_t)code * DFF + f] = f2bf(sact);
      }
    }
  }
}

// ---------------- FFN stage 2: y = act W2^T (K = 2048) ----------------
__global__ __launch_bounds__(256)
void ffn2_kernel(const u16* __restrict__ act, const u16* __restrict__ w2,
                 const int* __restrict__ cnt, const int* __restrict__ list,
                 float* __restrict__ y) {
  const int e = blockIdx.z;
  const int rows = cnt[e];
  const int m0 = blockIdx.x * 64;
  if (m0 >= rows) return;
  int nrows = rows - m0; if (nrows > 64) nrows = 64;
  const int d0 = blockIdx.y * 128;

  __shared__ __align__(16) u16 sA[64 * 32];
  __shared__ int srow[64];

  const int tid = threadIdx.x, lane = tid & 31, wid = tid >> 5;
  const int wm = wid & 3, wn = wid >> 2;

  if (tid < 64) srow[tid] = (tid < nrows) ? list[e * T_TOK + m0 + tid] : -1;

  v8f c[4];
#pragma unroll
  for (int n = 0; n < 4; n++)
#pragma unroll
    for (int j = 0; j < 8; j++) c[n][j] = 0.f;

  const u16* w2e = w2 + (size_t)e * DM * DFF;
  const int arow = tid >> 2;
  const int akq  = (tid & 3) * 8;

  for (int k0 = 0; k0 < DFF; k0 += 32) {
    __syncthreads();
    {
      int code = srow[arow];
      float4 v;
      if (code >= 0) v = *(const float4*)(act + (size_t)code * DFF + k0 + akq);
      else           v = make_float4(0.f, 0.f, 0.f, 0.f);
      *(float4*)&sA[arow * 32 + akq] = v;
    }
    __syncthreads();

    union { v16bf v; u16 u[16]; } af;
    const int ml = wm * 16 + (lane & 15);
    const int khalf = (lane & 16) ? 8 : 0;
#pragma unroll
    for (int v = 0; v < 8; v++) {
      int kk = 2 * v + 2 * khalf;
      af.u[2 * v + 0] = sA[ml * 32 + kk + 0];
      af.u[2 * v + 1] = sA[ml * 32 + kk + 1];
    }
    const int dcol = d0 + wn * 64 + (lane & 15);
    const int krow = k0 + ((lane & 16) ? 16 : 0);
#pragma unroll
    for (int n = 0; n < 4; n++) {
      const int d = dcol + n * 16;
      v16bf b = *(const v16bf*)(w2e + (size_t)d * DFF + krow);
      c[n] = __builtin_amdgcn_wmma_f32_16x16x32_bf16(false, af.v, false, b,
                                                     (short)0, c[n], false, false);
    }
  }

#pragma unroll
  for (int n = 0; n < 4; n++) {
#pragma unroll
    for (int j = 0; j < 8; j++) {
      int m = wm * 16 + j + ((lane & 16) ? 8 : 0);
      if (m < nrows) {
        int d = d0 + wn * 64 + n * 16 + (lane & 15);
        int code = srow[m];
        y[(size_t)code * DM + d] = c[n][j];
      }
    }
  }
}

// ---------------- combine: out[t] = s0*y[2t] + s1*y[2t+1] (deterministic) ----------------
__global__ void combine_kernel(const float* __restrict__ y, const float* __restrict__ score,
                               float* __restrict__ out) {
  int i = blockIdx.x * 256 + threadIdx.x;
  if (i < T_TOK * DM) {
    int t = i >> 9;
    int d = i & (DM - 1);
    out[i] = score[2 * t + 0] * y[(size_t)(2 * t + 0) * DM + d]
           + score[2 * t + 1] * y[(size_t)(2 * t + 1) * DM + d];
  }
}

extern "C" void kernel_launch(void* const* d_in, const int* in_sizes, int n_in,
                              void* d_out, int out_size, void* d_ws, size_t ws_size,
                              hipStream_t stream) {
  const float* x  = (const float*)d_in[0];
  const float* gw = (const float*)d_in[1];
  const float* W1 = (const float*)d_in[2];
  const float* V1 = (const float*)d_in[3];
  const float* W2 = (const float*)d_in[4];
  float* out = (float*)d_out;

  char* ws = (char*)d_ws;
  size_t o = 0;
  const size_t WB = (size_t)NE * DFF * DM * sizeof(u16);   // 16 MB each
  u16* w1b = (u16*)(ws + o); o += WB;
  u16* v1b = (u16*)(ws + o); o += WB;
  u16* w2b = (u16*)(ws + o); o += WB;
  u16* act = (u16*)(ws + o); o += (size_t)NR * DFF * sizeof(u16);   // 64 MB
  float* y = (float*)(ws + o); o += (size_t)NR * DM * sizeof(float); // 32 MB
  float* score = (float*)(ws + o); o += (size_t)NR * sizeof(float);
  int* idx  = (int*)(ws + o); o += (size_t)NR * sizeof(int);
  int* list = (int*)(ws + o); o += (size_t)NE * T_TOK * sizeof(int);
  int* cnt  = (int*)(ws + o); o += 128;
  float* psum = (float*)(ws + o); o += 32 * NE * sizeof(float);

  const int nW = NE * DFF * DM;   // 8M elements per weight tensor
  cvt_bf16_kernel<<<nW / 1024, 256, 0, stream>>>(W1, w1b, nW);
  cvt_bf16_kernel<<<nW / 1024, 256, 0, stream>>>(V1, v1b, nW);
  cvt_bf16_kernel<<<nW / 1024, 256, 0, stream>>>(W2, w2b, nW);

  init_cnt_kernel<<<1, 32, 0, stream>>>(cnt);
  gate_kernel<<<T_TOK / 256, 256, 0, stream>>>(x, gw, idx, score, psum);
  route_kernel<<<NR / 256, 256, 0, stream>>>(idx, cnt, list);
  aux_kernel<<<1, 32, 0, stream>>>(psum, cnt, out + (size_t)T_TOK * DM, T_TOK / 256);

  dim3 g1(T_TOK / 64, DFF / 128, NE);   // 128 x 16 x 8, blocks past cnt[e] exit
  ffn1_kernel<<<g1, 256, 0, stream>>>(x, w1b, v1b, cnt, list, act);

  dim3 g2(T_TOK / 64, DM / 128, NE);    // 128 x 4 x 8
  ffn2_kernel<<<g2, 256, 0, stream>>>(act, w2b, cnt, list, y);

  combine_kernel<<<(T_TOK * DM + 255) / 256, 256, 0, stream>>>(y, score, out);
}
